// RFEncoder_30090540876477
// MI455X (gfx1250) — matrix operand order repaired and checked by reading, not compiled, not run on hardware
//
#include <hip/hip_runtime.h>
#include <hip/hip_bf16.h>

typedef __attribute__((ext_vector_type(16))) _Float16 v16h;
typedef __attribute__((ext_vector_type(8)))  float    v8f;

#define HC 128          // HEADS * C
#define CDIM 64

#if defined(__gfx1250__) && __has_builtin(__builtin_amdgcn_global_load_async_to_lds_b128) && __has_builtin(__builtin_amdgcn_s_wait_asynccnt)
#define USE_ASYNC_LDS 1
typedef int v4i __attribute__((vector_size(16)));
typedef __attribute__((address_space(1))) v4i v4i_g;   // global
typedef __attribute__((address_space(3))) v4i v4i_l;   // LDS
#else
#define USE_ASYNC_LDS 0
#endif

__device__ __forceinline__ float lrelu(float v) { return v > 0.0f ? v : 0.2f * v; }

// Order-preserving float<->uint mapping so atomicMax(uint) == float max (incl. negatives).
__device__ __forceinline__ unsigned enc_ord(float f) {
    unsigned u = __float_as_uint(f);
    return (u >> 31) ? ~u : (u | 0x80000000u);
}
__device__ __forceinline__ float dec_ord(unsigned u) {
    return __uint_as_float((u & 0x80000000u) ? (u & 0x7FFFFFFFu) : ~u);
}

// ---------------- init kernels ----------------
__global__ void init_f32(float* p, float v, int n) {
    int i = blockIdx.x * blockDim.x + threadIdx.x;
    if (i < n) p[i] = v;
}
__global__ void init_u32(unsigned* p, unsigned v, int n) {
    int i = blockIdx.x * blockDim.x + threadIdx.x;
    if (i < n) p[i] = v;
}

// ---------------- layer-1 transform: fin = 2 ----------------
__global__ void gemm_fin2(const float* __restrict__ x, const float* __restrict__ W,
                          const float* __restrict__ b, float* __restrict__ out, int n) {
    int i = blockIdx.x * blockDim.x + threadIdx.x;
    if (i >= n * HC) return;
    int node = i >> 7, j = i & (HC - 1);
    float x0 = x[(size_t)node * 2 + 0], x1 = x[(size_t)node * 2 + 1];
    out[i] = fmaf(x0, W[j], fmaf(x1, W[HC + j], b[j]));
}

// ---------------- layer-2 transform: (N x 64) @ (64 x 128) + b, via WMMA ----------------
// Block = 256 threads (8 waves). Block computes a 16-row strip; wave w owns cols [16w,16w+16).
// Requires nrows % 16 == 0 (N = 100000 = 6250*16).
__global__ void __launch_bounds__(256)
gemm_wmma(const float* __restrict__ A, const float* __restrict__ W,
          const float* __restrict__ bias, float* __restrict__ out, int nrows) {
    __shared__ float At[16][68];   // 16 x 64 f32 tile; pad 4 -> fragment reads hit distinct banks
    const int row0 = blockIdx.x * 16;

#if USE_ASYNC_LDS
    {   // CDNA5 async copy: memory -> LDS directly, one B128 (16B) per thread, 4KB total
        int r  = threadIdx.x >> 4;        // tile row
        int ch = threadIdx.x & 15;        // 16B chunk within row
        const float* g = A + (size_t)(row0 + r) * 64 + ch * 4;
        __builtin_amdgcn_global_load_async_to_lds_b128(
            (v4i_g*)g, (v4i_l*)&At[r][ch * 4], 0, 0);
        __builtin_amdgcn_s_wait_asynccnt(0);
    }
    __syncthreads();
#else
    for (int i = threadIdx.x; i < 16 * 64; i += 256) {
        int r = i >> 6, k = i & 63;
        At[r][k] = A[(size_t)(row0 + r) * 64 + k];
    }
    __syncthreads();
#endif

    const int wave = threadIdx.x >> 5;
    const int lane = threadIdx.x & 31;
    const int col0 = wave * 16;
    const int n    = lane & 15;       // N index (and M index for A frag)
    const int hi   = lane >> 4;       // half-wave select
    const float bv = bias[col0 + n];

    v8f c = {};
    #pragma unroll
    for (int ks = 0; ks < 64; ks += 32) {
        // A fragment (16x32 f16): lane L: M = L%16; elems 0..7 -> K = kb+e, 8..15 -> K = 16+kb+e, kb = 8*(L>=16)
        v16h a, b;
        const int kbA = hi * 8;
        #pragma unroll
        for (int e = 0; e < 8; ++e) {
            a[e]     = (_Float16)At[n][ks + kbA + e];
            a[e + 8] = (_Float16)At[n][ks + 16 + kbA + e];
        }
        // B fragment (32x16 f16): lane L: N = L%16; elems 0..7 -> K = kb+e, 8..15 -> K = 16+kb+e, kb = 8*(L>=16)
        const int kbB = hi * 8;
        #pragma unroll
        for (int e = 0; e < 8; ++e) {
            b[e]     = (_Float16)W[(size_t)(ks + kbB + e) * HC + col0 + n];
            b[e + 8] = (_Float16)W[(size_t)(ks + 16 + kbB + e) * HC + col0 + n];
        }

        c = __builtin_amdgcn_wmma_f32_16x16x32_f16(false, a, false, b, (short)0, c, false, false);
    }

    // D layout: VGPR r -> row hi*8 + r, col = n. Branch-free epilogue (grid tiles exactly).
    #pragma unroll
    for (int r = 0; r < 8; ++r) {
        int row = row0 + hi * 8 + r;
        out[(size_t)row * HC + col0 + n] = c[r] + bv;
    }
}

// ---------------- edge pass A: alpha = <leakyrelu(xr[dst]+xl[src]), att>; segment max ----------------
__global__ void __launch_bounds__(256)
edge_alpha_max(const float* __restrict__ xl, const float* __restrict__ xr,
               const float* __restrict__ att, const int* __restrict__ ei,
               float* __restrict__ alpha, unsigned* __restrict__ m, int E) {
    int e = (blockIdx.x * blockDim.x + threadIdx.x) >> 5;
    if (e >= E) return;
    int lane = threadIdx.x & 31;
    int head = lane >> 4;
    int l    = lane & 15;
    int src = ei[e], dst = ei[E + e];

    const float4* xs = (const float4*)(xl + (size_t)src * HC + head * CDIM);
    const float4* xd = (const float4*)(xr + (size_t)dst * HC + head * CDIM);
    const float4* at = (const float4*)(att + head * CDIM);
    float4 s = xs[l], d = xd[l], a4 = at[l];
    float sum = lrelu(s.x + d.x) * a4.x + lrelu(s.y + d.y) * a4.y +
                lrelu(s.z + d.z) * a4.z + lrelu(s.w + d.w) * a4.w;
    #pragma unroll
    for (int off = 8; off >= 1; off >>= 1) sum += __shfl_xor(sum, off, 32);
    if (l == 0) {
        alpha[(size_t)e * 2 + head] = sum;
        atomicMax(&m[(size_t)dst * 2 + head], enc_ord(sum));
    }
}

// ---------------- edge pass B: a = exp(alpha - m[dst]); segment sum ----------------
__global__ void edge_exp_sum(float* __restrict__ alpha, const unsigned* __restrict__ m,
                             float* __restrict__ denom, const int* __restrict__ ei, int E) {
    int i = blockIdx.x * blockDim.x + threadIdx.x;
    if (i >= E * 2) return;
    int e = i >> 1, head = i & 1;
    int dst = ei[E + e];
    float a = __expf(alpha[i] - dec_ord(m[(size_t)dst * 2 + head]));
    alpha[i] = a;
    atomicAdd(&denom[(size_t)dst * 2 + head], a);
}

// ---------------- edge pass C: acc[dst] += 0.25 * sum_h (a_h/denom_h) * xl[src,h,:] ----------------
__global__ void __launch_bounds__(256)
edge_scatter(const float* __restrict__ xl, const float* __restrict__ alpha,
             const float* __restrict__ denom, const int* __restrict__ ei,
             float* __restrict__ acc, int E) {
    int e = (blockIdx.x * blockDim.x + threadIdx.x) >> 5;
    if (e >= E) return;
    int lane = threadIdx.x & 31;
    int src = ei[e], dst = ei[E + e];
    float w0 = alpha[(size_t)e * 2 + 0] / (denom[(size_t)dst * 2 + 0] + 1e-16f);
    float w1 = alpha[(size_t)e * 2 + 1] / (denom[(size_t)dst * 2 + 1] + 1e-16f);
    const float2* x0 = (const float2*)(xl + (size_t)src * HC) + lane;
    const float2* x1 = (const float2*)(xl + (size_t)src * HC + CDIM) + lane;
    float2 a = *x0, b = *x1;
    float v0 = 0.25f * (w0 * a.x + w1 * b.x);
    float v1 = 0.25f * (w0 * a.y + w1 * b.y);
    atomicAdd(&acc[(size_t)dst * CDIM + lane * 2 + 0], v0);
    atomicAdd(&acc[(size_t)dst * CDIM + lane * 2 + 1], v1);
}

// ---------------- finalize: out = [relu](acc + 0.5*(b1 + b2)) ----------------
__global__ void finalize(float* __restrict__ acc, const float* __restrict__ b1,
                         const float* __restrict__ b2, int n, int do_relu) {
    int i = blockIdx.x * blockDim.x + threadIdx.x;
    if (i >= n) return;
    int c = i & (CDIM - 1);
    float v = acc[i] + 0.5f * (b1[c] + b2[c]);
    if (do_relu) v = fmaxf(v, 0.0f);
    acc[i] = v;
}

extern "C" void kernel_launch(void* const* d_in, const int* in_sizes, int n_in,
                              void* d_out, int out_size, void* d_ws, size_t ws_size,
                              hipStream_t stream) {
    const float* x      = (const float*)d_in[0];
    const int*   ei_adj = (const int*)d_in[1];
    const int*   ei_ray = (const int*)d_in[2];
    const int N = in_sizes[0] / 2;       // IN_DIM = 2
    const int E = in_sizes[1] / 2;       // ei shape (2, E)

    auto P = [&](int base, int k) { return (const float*)d_in[base + k]; };

    char* ws = (char*)d_ws;
    size_t off = 0;
    auto carve = [&](size_t bytes) { void* p = ws + off; off = (off + bytes + 255) & ~(size_t)255; return p; };
    float*    xl    = (float*)carve((size_t)N * HC * 4);
    float*    xr    = (float*)carve((size_t)N * HC * 4);
    float*    alpha = (float*)carve((size_t)E * 2 * 4);
    unsigned* m     = (unsigned*)carve((size_t)N * 2 * 4);
    float*    denom = (float*)carve((size_t)N * 2 * 4);
    float*    h     = (float*)carve((size_t)N * CDIM * 4);
    float*    out   = (float*)d_out;

    const int T = 256;
    auto blk = [](long n, int t) { return (int)((n + t - 1) / t); };
    const unsigned ENC_NEG_INF = 0x007FFFFFu;   // enc_ord(-inf)

    for (int layer = 0; layer < 2; ++layer) {
        float* acc = layer ? out : h;
        init_f32<<<blk((long)N * CDIM, T), T, 0, stream>>>(acc, 0.0f, N * CDIM);

        const int  bases[2] = { layer ? 15 : 3, layer ? 21 : 9 };
        const int* eis[2]   = { ei_adj, ei_ray };
        const float* in_feat = layer ? h : x;

        for (int g = 0; g < 2; ++g) {
            const float* Wl = P(bases[g], 0), *bl = P(bases[g], 1);
            const float* Wr = P(bases[g], 2), *br = P(bases[g], 3);
            const float* at = P(bases[g], 4);
            const int*   ei = eis[g];

            if (layer == 0) {
                gemm_fin2<<<blk((long)N * HC, T), T, 0, stream>>>(x, Wl, bl, xl, N);
                gemm_fin2<<<blk((long)N * HC, T), T, 0, stream>>>(x, Wr, br, xr, N);
            } else {
                gemm_wmma<<<N / 16, T, 0, stream>>>(in_feat, Wl, bl, xl, N);
                gemm_wmma<<<N / 16, T, 0, stream>>>(in_feat, Wr, br, xr, N);
            }

            init_u32<<<blk((long)N * 2, T), T, 0, stream>>>(m, ENC_NEG_INF, N * 2);
            init_f32<<<blk((long)N * 2, T), T, 0, stream>>>(denom, 0.0f, N * 2);

            edge_alpha_max<<<blk((long)E * 32, T), T, 0, stream>>>(xl, xr, at, ei, alpha, m, E);
            edge_exp_sum  <<<blk((long)E * 2,  T), T, 0, stream>>>(alpha, m, denom, ei, E);
            edge_scatter  <<<blk((long)E * 32, T), T, 0, stream>>>(xl, alpha, denom, ei, acc, E);
        }
        finalize<<<blk((long)N * CDIM, T), T, 0, stream>>>(acc, P(bases[0], 5), P(bases[1], 5),
                                                           N * CDIM, layer == 0 ? 1 : 0);
    }
}
